// EdgeModel_11227044512371
// MI455X (gfx1250) — matrix operand order repaired and checked by reading, not compiled, not run on hardware
//
#include <hip/hip_runtime.h>
#include <hip/hip_bf16.h>

// ---------------------------------------------------------------------------
// EdgeModel on MI455X (gfx1250, wave32, WMMA 16x16x32 f16 -> f32)
//
// a = mlp3(step_x) @ post_w1   (f16, [1024][64] in ws)
// b = mlp3(state_x)@ post_w1   (f16, [1024][64] in ws)
// pair(i,j): h1 = relu(a[i]+b[j]+post_b1); h2 = relu(h1@w2+b2); out = h2@w3+b3
// then mutual-argmax matching on sigmoid(logits_st[...,0]).
// ---------------------------------------------------------------------------

typedef __attribute__((ext_vector_type(16))) _Float16 v16h;
typedef __attribute__((ext_vector_type(8)))  _Float16 half8;
typedef __attribute__((ext_vector_type(8)))  float    v8f;

#define LSTR 72   // LDS row stride in f16 elements (144B, 16B-aligned rows)

__device__ __forceinline__ int kmapA(int e, int half) {
  // ISA 7.12.2, 16-bit A 16x32: lanes 0-15: elems0-7 -> K0-7, elems8-15 -> K16-23
  //                             lanes16-31: elems0-7 -> K8-15, elems8-15 -> K24-31
  return (e < 8) ? (8 * half + e) : (8 + 8 * half + e);
}

// A fragment = two contiguous 8xf16 runs per lane -> two b128 loads.
__device__ __forceinline__ v16h load_afrag128(const _Float16* m, int lane,
                                              int kbase) {
  int row = lane & 15, half = lane >> 4;
  const half8* p0 = (const half8*)(m + row * LSTR + kbase + 8 * half);
  const half8* p1 = (const half8*)(m + row * LSTR + kbase + 16 + 8 * half);
  half8 lo = *p0, hi = *p1;
  return __builtin_shufflevector(lo, hi, 0, 1, 2, 3, 4, 5, 6, 7, 8, 9, 10, 11,
                                 12, 13, 14, 15);
}

__device__ __forceinline__ v16h load_bfrag(const _Float16* m, int ldb,
                                           int lane, int kbase, int nbase) {
  // dense 32x16 B: lanes0-15 elems0-7 K0-7, elems8-15 K16-23; lanes16-31 +8
  int n = nbase + (lane & 15), half = lane >> 4;
  v16h f;
#pragma unroll
  for (int e = 0; e < 16; ++e)
    f[e] = m[(kbase + 8 * half + e + (e & 8)) * ldb + n];
  return f;
}

__device__ __forceinline__ v8f wmma_f16(v16h a, v16h b, v8f c) {
  return __builtin_amdgcn_wmma_f32_16x16x32_f16(
      false, a, false, b, (short)0, c, false, false);
}

// ------------------------- Phase 1: pre-MLP + W1 projection -----------------

template <int DO_RELU>
__device__ __forceinline__ void mlp_layer16(
    const float* __restrict__ wg, const float* __restrict__ bg, float bscale,
    _Float16 (&wl)[64 * 64], _Float16 (&hin)[16 * LSTR],
    _Float16 (&hout)[16 * LSTR], int lane) {
  // stage 64x64 weight f32 -> f16 (same-wave LDS, in-order)
  for (int idx = lane; idx < 64 * 64; idx += 32)
    wl[idx] = (_Float16)wg[idx];

  v16h a0 = load_afrag128(hin, lane, 0);
  v16h a1 = load_afrag128(hin, lane, 32);
  int n0 = lane & 15, half = lane >> 4;
#pragma unroll
  for (int t = 0; t < 4; ++t) {
    float bias = bg[t * 16 + n0] * bscale;   // always-valid ptr, no branch
    v8f c;
#pragma unroll
    for (int r = 0; r < 8; ++r) c[r] = bias; // bias folded into accumulator
    c = wmma_f16(a0, load_bfrag(wl, 64, lane, 0, t * 16), c);
    c = wmma_f16(a1, load_bfrag(wl, 64, lane, 32, t * 16), c);
#pragma unroll
    for (int r = 0; r < 8; ++r) {
      float v = c[r];
      if (DO_RELU) v = v > 0.f ? v : 0.f;
      // D layout: lane half=0 -> M=r, half=1 -> M=r+8; N = lane&15
      hout[(r + 8 * half) * LSTR + t * 16 + n0] = (_Float16)v;
    }
  }
}

__global__ __launch_bounds__(32) void pre_mlp_kernel(
    const float* __restrict__ step_x, const float* __restrict__ state_x,
    const float* __restrict__ w1, const float* __restrict__ b1,
    const float* __restrict__ w2, const float* __restrict__ b2,
    const float* __restrict__ w3, const float* __restrict__ b3,
    const float* __restrict__ q1,  // post_w1
    _Float16* __restrict__ a_all)  // [2048][64]: step rows, then state rows
{
  __shared__ __align__(16) _Float16 wlds[64 * 64];
  __shared__ __align__(16) _Float16 buf0[16 * LSTR];
  __shared__ __align__(16) _Float16 buf1[16 * LSTR];

  int lane = threadIdx.x;
  int blk = blockIdx.x;                 // 0..127
  const float* src;
  int row_out;
  if (blk < 64) { src = step_x + blk * 16 * 64;          row_out = blk * 16; }
  else          { src = state_x + (blk - 64) * 16 * 64;  row_out = 1024 + (blk - 64) * 16; }

  // stage input tile 16x64 f32 -> f16
  for (int idx = lane; idx < 16 * 64; idx += 32) {
    int r = idx >> 6, c = idx & 63;
    buf0[r * LSTR + c] = (_Float16)src[r * 64 + c];
  }

  mlp_layer16<1>(w1, b1, 1.f, wlds, buf0, buf1, lane);  // relu(x@w1+b1)
  mlp_layer16<1>(w2, b2, 1.f, wlds, buf1, buf0, lane);  // relu(h1@w2+b2)
  mlp_layer16<0>(w3, b3, 1.f, wlds, buf0, buf1, lane);  // h2@w3+b3
  mlp_layer16<0>(q1, b3, 0.f, wlds, buf1, buf0, lane);  // sx@post_w1 (no bias)

  for (int idx = lane; idx < 16 * 64; idx += 32) {
    int r = idx >> 6, c = idx & 63;
    a_all[(row_out + r) * 64 + c] = buf0[r * LSTR + c];
  }
}

// ------------------------- Phase 2: pairwise post-MLP -----------------------

#define ITER_I 32   // i values per wave; gridDim.y * ITER_I == 1024

__global__ __launch_bounds__(128) void pair_mlp_kernel(
    const _Float16* __restrict__ a_all,  // [2048][64]
    const float* __restrict__ qb1, const float* __restrict__ qw2,
    const float* __restrict__ qb2, const float* __restrict__ qw3,
    const float* __restrict__ qb3,
    float* __restrict__ out_ss, float* __restrict__ out_st)
{
  __shared__ __align__(16) _Float16 w2lds[64 * 64];
  __shared__ __align__(16) _Float16 btile[4][16 * LSTR];
  __shared__ __align__(16) _Float16 h2lds[4][16 * LSTR];

  int tid = threadIdx.x, lane = tid & 31, wave = tid >> 5;

  // cooperative stage of post_w2 (f32 -> f16)
  for (int idx = tid; idx < 64 * 64; idx += 128)
    w2lds[idx] = (_Float16)qw2[idx];
  __syncthreads();

  int jblock = blockIdx.x * 4 + wave;        // 0..63
  int j0 = jblock * 16;
  int i0 = blockIdx.y * ITER_I;
  const _Float16* aP = a_all;                               // step projections
  const _Float16* bP = (blockIdx.z == 0) ? a_all : (a_all + 1024 * 64);
  float* outP = (blockIdx.z == 0) ? out_ss : out_st;

  // stage this wave's b tile [16][64] (same-wave LDS, in-order, no barrier)
  _Float16* bt = &btile[wave][0];
  for (int idx = lane; idx < 16 * 64; idx += 32) {
    int r = idx >> 6, c = idx & 63;
    bt[r * LSTR + c] = bP[(j0 + r) * 64 + c];
  }

  int half = lane >> 4, mrow = lane & 15, n0 = lane & 15;

  // per-lane (b[j] + post_b1) in A-fragment element order, packed f16
  v16h bb0, bb1;
#pragma unroll
  for (int e = 0; e < 16; ++e) {
    int k = kmapA(e, half);
    bb0[e] = (_Float16)((float)bt[mrow * LSTR + k] + qb1[k]);
    bb1[e] = (_Float16)((float)bt[mrow * LSTR + k + 32] + qb1[k + 32]);
  }

  // post_w2 B fragments (2 K-halves x 4 N-tiles), built once per wave
  v16h bw2[2][4];
#pragma unroll
  for (int kb = 0; kb < 2; ++kb)
#pragma unroll
    for (int t = 0; t < 4; ++t)
      bw2[kb][t] = load_bfrag(w2lds, 64, lane, kb * 32, t * 16);

  // post_w3 [64][2] zero-padded to 16 columns
  v16h bw3[2];
#pragma unroll
  for (int kb = 0; kb < 2; ++kb)
#pragma unroll
    for (int e = 0; e < 16; ++e) {
      int k = kb * 32 + 8 * half + e + (e & 8);
      bw3[kb][e] = (n0 < 2) ? (_Float16)qw3[k * 2 + n0] : (_Float16)0.f;
    }

  float bias2[4];
#pragma unroll
  for (int t = 0; t < 4; ++t) bias2[t] = qb2[t * 16 + n0];
  float bias3 = (n0 < 2) ? qb3[n0] : 0.f;

  _Float16* h2 = &h2lds[wave][0];
  v16h hzero = {};

  for (int ii = 0; ii < ITER_I; ++ii) {
    int i = i0 + ii;
    // a[i] chunks straight from global (L2-resident, 4x global_load_b128)
    const _Float16* arg = aP + (long)i * 64;
    half8 c0 = *(const half8*)(arg + 8 * half);
    half8 c1 = *(const half8*)(arg + 16 + 8 * half);
    half8 c2 = *(const half8*)(arg + 32 + 8 * half);
    half8 c3 = *(const half8*)(arg + 48 + 8 * half);
    if (ii + 1 < ITER_I)
      __builtin_prefetch(aP + (long)(i + 1) * 64, 0, 1);  // global_prefetch_b8

    // h1 = relu(a + (b + b1)) in packed f16 (v_pk_add / v_pk_max)
    v16h a0 = __builtin_shufflevector(c0, c1, 0, 1, 2, 3, 4, 5, 6, 7, 8, 9,
                                      10, 11, 12, 13, 14, 15) + bb0;
    v16h a1 = __builtin_shufflevector(c2, c3, 0, 1, 2, 3, 4, 5, 6, 7, 8, 9,
                                      10, 11, 12, 13, 14, 15) + bb1;
    a0 = __builtin_elementwise_max(a0, hzero);
    a1 = __builtin_elementwise_max(a1, hzero);

    // layer 2: h2 = relu(h1 @ w2 + b2), bias folded into accumulator
#pragma unroll
    for (int t = 0; t < 4; ++t) {
      v8f c;
#pragma unroll
      for (int r = 0; r < 8; ++r) c[r] = bias2[t];
      c = wmma_f16(a0, bw2[0][t], c);
      c = wmma_f16(a1, bw2[1][t], c);
#pragma unroll
      for (int r = 0; r < 8; ++r) {
        float v = c[r];
        h2[(r + 8 * half) * LSTR + t * 16 + n0] =
            (_Float16)(v > 0.f ? v : 0.f);
      }
    }

    // layer 3: out = h2 @ w3pad + b3 (columns 0,1 valid)
    v16h h0 = load_afrag128(h2, lane, 0);
    v16h h1 = load_afrag128(h2, lane, 32);
    v8f o;
#pragma unroll
    for (int r = 0; r < 8; ++r) o[r] = bias3;
    o = wmma_f16(h0, bw3[0], o);
    o = wmma_f16(h1, bw3[1], o);

    if (n0 < 2) {
#pragma unroll
      for (int r = 0; r < 8; ++r) {
        int j = j0 + r + 8 * half;
        outP[((long)i * 1024 + j) * 2 + n0] = o[r];
      }
    }
  }
}

// ------------------------- Phase 3: mutual-argmax matching ------------------

__global__ void col_argmax_kernel(const float* __restrict__ logits_st,
                                  int* __restrict__ col_arg) {
  int j = blockIdx.x * blockDim.x + threadIdx.x;
  if (j >= 1024) return;
  float best = logits_st[(long)j * 2];
  int bi = 0;
  for (int i = 1; i < 1024; ++i) {
    float v = logits_st[((long)i * 1024 + j) * 2];
    if (v > best) { best = v; bi = i; }    // strict > keeps first index
  }
  col_arg[j] = bi;
}

__global__ __launch_bounds__(256) void row_match_kernel(
    const float* __restrict__ logits_st, const int* __restrict__ col_arg,
    float* __restrict__ matched) {
  int i = blockIdx.x * 8 + (threadIdx.x >> 5);
  int lane = threadIdx.x & 31;

  float best = -1.f;
  int bj = 0;
  for (int j = lane; j < 1024; j += 32) {
    float s = 0.f;
    if (col_arg[j] == i) {
      float x = logits_st[((long)i * 1024 + j) * 2];
      s = 1.f / (1.f + __expf(-x));
    }
    if (s > best) { best = s; bj = j; }
  }
  // wave32 argmax reduction, ties -> smallest j (matches jnp.argmax)
#pragma unroll
  for (int m = 16; m >= 1; m >>= 1) {
    float ob = __shfl_xor(best, m, 32);
    int oj = __shfl_xor(bj, m, 32);
    if (ob > best || (ob == best && oj < bj)) { best = ob; bj = oj; }
  }

  for (int j = lane; j < 1024; j += 32)
    matched[(long)i * 1024 + j] = (j == bj) ? best : 0.f;
}

// ------------------------- launch -------------------------------------------

extern "C" void kernel_launch(void* const* d_in, const int* in_sizes, int n_in,
                              void* d_out, int out_size, void* d_ws,
                              size_t ws_size, hipStream_t stream) {
  const float* step_x  = (const float*)d_in[0];
  const float* state_x = (const float*)d_in[1];
  const float* pre_w1  = (const float*)d_in[2];
  const float* pre_b1  = (const float*)d_in[3];
  const float* pre_w2  = (const float*)d_in[4];
  const float* pre_b2  = (const float*)d_in[5];
  const float* pre_w3  = (const float*)d_in[6];
  const float* pre_b3  = (const float*)d_in[7];
  const float* post_w1 = (const float*)d_in[8];
  const float* post_b1 = (const float*)d_in[9];
  const float* post_w2 = (const float*)d_in[10];
  const float* post_b2 = (const float*)d_in[11];
  const float* post_w3 = (const float*)d_in[12];
  const float* post_b3 = (const float*)d_in[13];

  float* out     = (float*)d_out;
  float* out_ss  = out;                         // [1024][1024][2]
  float* out_st  = out + 2 * 1024 * 1024;       // [1024][1024][2]
  float* matched = out + 4 * 1024 * 1024;       // [1024][1024]

  _Float16* a_all = (_Float16*)d_ws;            // [2048][64] f16 (256 KB)
  int* col_arg = (int*)((char*)d_ws + (size_t)2048 * 64 * sizeof(_Float16));

  // Phase 1: a = mlp3(step)@post_w1 ; b = mlp3(state)@post_w1
  pre_mlp_kernel<<<128, 32, 0, stream>>>(step_x, state_x, pre_w1, pre_b1,
                                         pre_w2, pre_b2, pre_w3, pre_b3,
                                         post_w1, a_all);

  // Phase 2: 2M pair rows through layers 2+3 via WMMA
  dim3 g2(16, 1024 / ITER_I, 2);
  pair_mlp_kernel<<<g2, 128, 0, stream>>>(a_all, post_b1, post_w2, post_b2,
                                          post_w3, post_b3, out_ss, out_st);

  // Phase 3: mutual-argmax matching on sigmoid(logits_st[...,0])
  col_argmax_kernel<<<4, 256, 0, stream>>>(out_st, col_arg);
  row_match_kernel<<<128, 256, 0, stream>>>(out_st, col_arg, matched);
}
